// MultiHeadAttention_79886391706295
// MI455X (gfx1250) — matrix operand order repaired and checked by reading, not compiled, not run on hardware
//
#include <hip/hip_runtime.h>
#include <math.h>

#define BB 4
#define LL 1024
#define DD 1024
#define HH 16
#define DKK 64

typedef __bf16 bf16_t;
typedef __attribute__((ext_vector_type(16))) bf16_t v16bf;
typedef __attribute__((ext_vector_type(8)))  bf16_t v8bf;
typedef __attribute__((ext_vector_type(8)))  float  v8f;
typedef __attribute__((ext_vector_type(4)))  float  vf4;
typedef __attribute__((ext_vector_type(4)))  unsigned int v4u;
typedef __attribute__((ext_vector_type(8)))  int v8i;
typedef __attribute__((ext_vector_type(4)))  int v4i;

// D = A(16x32 bf16) * B(32x16 bf16) + C(16x16 f32) -- v_wmma_f32_16x16x32_bf16
static __device__ __forceinline__ v8f wmma_bf16(v16bf a, v16bf b, v8f c) {
  return __builtin_amdgcn_wmma_f32_16x16x32_bf16(false, a, false, b, (short)0, c,
                                                 false, false);
}

static __device__ __forceinline__ v16bf cat16(v8bf lo, v8bf hi) {
  return __builtin_shufflevector(lo, hi, 0, 1, 2, 3, 4, 5, 6, 7, 8, 9, 10, 11,
                                 12, 13, 14, 15);
}

// load 8 contiguous f32 (global or LDS), convert to bf16
static __device__ __forceinline__ v8bf cvt8(const float* p) {
  vf4 f0 = *(const vf4*)p;
  vf4 f1 = *(const vf4*)(p + 4);
  v8bf r;
  r[0] = (bf16_t)f0[0]; r[1] = (bf16_t)f0[1];
  r[2] = (bf16_t)f0[2]; r[3] = (bf16_t)f0[3];
  r[4] = (bf16_t)f1[0]; r[5] = (bf16_t)f1[1];
  r[6] = (bf16_t)f1[2]; r[7] = (bf16_t)f1[3];
  return r;
}

static __device__ __forceinline__ v8bf ld8bf(const bf16_t* __restrict__ p) {
  return *(const v8bf*)p;
}

// A operand (16 rows x 32 K) from row-major f32, ld = row stride (elements).
// lanes 0-15: K=0..7,16..23 ; lanes 16-31: K=8..15,24..31 (CDNA5 16-bit A layout)
static __device__ __forceinline__ v16bf loadA_lds_f32(const float* X, int ld,
                                                      int m0, int lane) {
  int m = lane & 15, ko = (lane >> 4) * 8;
  const float* p = X + (m0 + m) * ld + ko;
  return cat16(cvt8(p), cvt8(p + 16));
}

// B operand (32x16) where B[k][n] = SRC[n0+n][k] (rows of SRC are K-contiguous)
// lanes 0-15: col n=lane, K=0..7,16..23 ; lanes 16-31: col n=lane-16, K=8..15,24..31
static __device__ __forceinline__ v16bf loadB_lds_f32(const float* W, int ld,
                                                      int n0, int lane) {
  int n = lane & 15, ko = (lane >> 4) * 8;
  const float* p = W + (n0 + n) * ld + ko;
  return cat16(cvt8(p), cvt8(p + 16));
}

static __device__ __forceinline__ v16bf loadA_bf(const bf16_t* __restrict__ X,
                                                 int ld, int m0, int kb, int lane) {
  int m = lane & 15, ko = (lane >> 4) * 8;
  const bf16_t* p = X + (size_t)(m0 + m) * ld + kb + ko;
  return cat16(ld8bf(p), ld8bf(p + 16));
}

static __device__ __forceinline__ v16bf loadB_rows_bf(const bf16_t* __restrict__ W,
                                                      int ld, int n0, int kb, int lane) {
  int n = lane & 15, ko = (lane >> 4) * 8;
  const bf16_t* p = W + (size_t)(n0 + n) * ld + kb + ko;
  return cat16(ld8bf(p), ld8bf(p + 16));
}

// ---- Tensor Data Mover: DMA a 2D tile (tile0 x tile1 f32) into LDS.
// Builds D# group0/group1 per CDNA5 ISA 8.3/8.4 (data_size=4B, type=2).
// lds_off is the byte offset of the destination inside the block's LDS
// allocation (single __shared__ allocation -> starts at offset 0).
// This toolchain exposes the 6-arg builtin: (g0, g1, g2, g3, g4, cpol) where
// g4 is the unused 5th descriptor slot (VADDR4 = NULL) -> zero-filled.
static __device__ __forceinline__ void tdm_load_2d(unsigned int lds_off,
                                                   const float* gaddr,
                                                   unsigned int dim0,
                                                   unsigned int dim1,
                                                   unsigned int tile0,
                                                   unsigned int tile1,
                                                   unsigned int stride0) {
  unsigned long long ga = (unsigned long long)(uintptr_t)gaddr;
  v4u g0;
  g0[0] = 1u;                                     // count=1, user mode
  g0[1] = lds_off;                                // lds_addr
  g0[2] = (unsigned int)ga;                       // global_addr[31:0]
  g0[3] = (unsigned int)(ga >> 32) | 0x80000000u; // global_addr[56:32] | type=2
  v8i g1;
  g1[0] = (int)(2u << 16);                        // data_size=4B; no mask/flags
  g1[1] = (int)(dim0 << 16);                      // tensor_dim0[15:0] in [31:16]
  g1[2] = (int)((dim0 >> 16) | (dim1 << 16));     // dim0 hi | dim1 lo
  g1[3] = (int)((dim1 >> 16) | (tile0 << 16));    // dim1 hi | tile_dim0
  g1[4] = (int)tile1;                             // tile_dim1 (tile_dim2=0)
  g1[5] = (int)stride0;                           // tensor_dim0_stride[31:0]
  g1[6] = 0;                                      // stride0 hi | stride1 lo
  g1[7] = 0;
  v4i g2 = {0, 0, 0, 0};                          // 2D: dims2/3 unused
  v4i g3 = {0, 0, 0, 0};
  v8i g4 = {0, 0, 0, 0, 0, 0, 0, 0};              // unused slot (NULL)
  __builtin_amdgcn_tensor_load_to_lds(g0, g1, g2, g3, g4, 0);
}

// ---------------- Projection: dst = X[4096x1024] @ W^T, LDS-staged via TDM.
// Block = 256 thr (8 waves) computes a 128x128 output region; K-slabs of 32
// double-buffered in LDS (A:128x32 f32, B:128x32 f32), loaded by the TDM and
// synchronized with s_wait_tensorcnt + barrier. Each wave: 32(M)x64(N),
// 8 f32 accumulators, 8 WMMAs per K-step.
// mode 0/1: dst[b][h][l][dk] (bf16), mode 2: dst[b][h][dv][l] (V transposed)
__global__ __launch_bounds__(256) void proj_kernel(const float* __restrict__ X,
                                                   const float* __restrict__ W,
                                                   bf16_t* __restrict__ dst,
                                                   int mode, float scale) {
  __shared__ __align__(16) unsigned char smem[65536]; // A0,A1,B0,B1 16KB each
  int lane = threadIdx.x & 31;
  int wave = threadIdx.x >> 5;
  int m0 = (blockIdx.x >> 3) * 128;   // 32 M-blocks
  int n0 = (blockIdx.x & 7) * 128;    // 8 N-blocks
  int wm = wave >> 1;                 // 0..3 : 32-row strip
  int wn = wave & 1;                  // 0..1 : 64-col strip

  v8f acc[8];
#pragma unroll
  for (int i = 0; i < 8; ++i) acc[i] = (v8f){};

  const int NSTEP = DD / 32;          // 32 K-steps
  if (wave == 0) {                    // prologue: fill buffer 0
    tdm_load_2d(0u,      X + (size_t)m0 * DD, DD, 4096, 32, 128, DD);
    tdm_load_2d(32768u,  W + (size_t)n0 * DD, DD, 1024, 32, 128, DD);
  }

  for (int s = 0; s < NSTEP; ++s) {
    if (wave == 0) __builtin_amdgcn_s_wait_tensorcnt(0);
    __syncthreads();                  // slab s ready; all done reading s-1
    if (wave == 0 && s + 1 < NSTEP) { // prefetch next slab into other buffer
      unsigned int b = (unsigned int)((s + 1) & 1) * 16384u;
      int kb = (s + 1) * 32;
      tdm_load_2d(b,          X + (size_t)m0 * DD + kb, DD, 4096, 32, 128, DD);
      tdm_load_2d(32768u + b, W + (size_t)n0 * DD + kb, DD, 1024, 32, 128, DD);
    }
    const float* As = (const float*)(smem + (unsigned)(s & 1) * 16384u);
    const float* Bs = (const float*)(smem + 32768u + (unsigned)(s & 1) * 16384u);
    v16bf a0 = loadA_lds_f32(As, 32, wm * 32, lane);
    v16bf a1 = loadA_lds_f32(As, 32, wm * 32 + 16, lane);
#pragma unroll
    for (int tn = 0; tn < 4; ++tn) {
      v16bf b = loadB_lds_f32(Bs, 32, wn * 64 + tn * 16, lane);
      acc[tn]     = wmma_bf16(a0, b, acc[tn]);
      acc[4 + tn] = wmma_bf16(a1, b, acc[4 + tn]);
    }
  }

  int col = lane & 15, hi = lane >> 4;
#pragma unroll
  for (int tm = 0; tm < 2; ++tm) {
#pragma unroll
    for (int tn = 0; tn < 4; ++tn) {
      v8f c = acc[tm * 4 + tn];
      int nglob = n0 + wn * 64 + tn * 16 + col;
      int h = nglob >> 6, dk = nglob & 63;
#pragma unroll
      for (int r = 0; r < 8; ++r) {
        int row = m0 + wm * 32 + tm * 16 + r + 8 * hi;  // [0,4096)
        int bidx = row >> 10, l = row & 1023;
        bf16_t val = (bf16_t)(c[r] * scale);
        if (mode == 2)
          dst[(((size_t)(bidx * HH + h) * DKK + dk) << 10) + l] = val;
        else
          dst[(((size_t)(bidx * HH + h) << 10) + l) * DKK + dk] = val;
      }
    }
  }
}

// ---------------- Attention: per-wave 16-row q tile, two-pass causal softmax
__global__ __launch_bounds__(256) void attn_kernel(const bf16_t* __restrict__ qh,
                                                   const bf16_t* __restrict__ kh,
                                                   const bf16_t* __restrict__ vT,
                                                   float* __restrict__ attn,
                                                   float* __restrict__ o_ws) {
  __shared__ bf16_t pbuf[8][16 * 32];   // per-wave 16x32 P staging (A-layout feed)
  int lane = threadIdx.x & 31;
  int wave = threadIdx.x >> 5;
  int bh = blockIdx.x >> 3;                    // 0..63  (b*16+h)
  int qt = (blockIdx.x & 7) * 8 + wave;        // 0..63  q-tile index
  int qbase = qt * 16;

  const bf16_t* Q = qh + (size_t)bh * LL * DKK;
  const bf16_t* K = kh + (size_t)bh * LL * DKK;
  const bf16_t* V = vT + (size_t)bh * DKK * LL;
  float* A = attn + (size_t)bh * LL * LL;

  v16bf a0 = loadA_bf(Q, DKK, qbase, 0, lane);
  v16bf a1 = loadA_bf(Q, DKK, qbase, 32, lane);

  int col = lane & 15, hi = lane >> 4;

  float m[8], l[8];
#pragma unroll
  for (int r = 0; r < 8; ++r) { m[r] = -3.0e38f; l[r] = 0.0f; }

  // ---- pass 1: online max / sum over causal key tiles
  for (int kt = 0; kt <= qt; ++kt) {
    int n0 = kt * 16;
    v16bf b0 = loadB_rows_bf(K, DKK, n0, 0, lane);
    v16bf b1 = loadB_rows_bf(K, DKK, n0, 32, lane);
    v8f s = {};
    s = wmma_bf16(a0, b0, s);
    s = wmma_bf16(a1, b1, s);
    int kj = n0 + col;
#pragma unroll
    for (int r = 0; r < 8; ++r) {
      int qi = qbase + r + 8 * hi;
      float sv = (kj <= qi) ? s[r] : -1e9f;
      float mx = sv;                       // row max within 16-lane half
      mx = fmaxf(mx, __shfl_xor(mx, 1, 32));
      mx = fmaxf(mx, __shfl_xor(mx, 2, 32));
      mx = fmaxf(mx, __shfl_xor(mx, 4, 32));
      mx = fmaxf(mx, __shfl_xor(mx, 8, 32));
      float mn = fmaxf(m[r], mx);
      float p = __expf(sv - mn);
      p += __shfl_xor(p, 1, 32);
      p += __shfl_xor(p, 2, 32);
      p += __shfl_xor(p, 4, 32);
      p += __shfl_xor(p, 8, 32);
      l[r] = l[r] * __expf(m[r] - mn) + p;
      m[r] = mn;
    }
  }
  float rinv[8];
#pragma unroll
  for (int r = 0; r < 8; ++r) rinv[r] = 1.0f / l[r];

  // ---- pass 2: P = softmax(S) -> attn out + O = P @ V
  v8f o0 = {}, o1 = {}, o2 = {}, o3 = {};
  bf16_t* myP = &pbuf[wave][0];
  int nk = (qt + 1) * 16;                // valid key columns
  for (int kk = 0; kk < nk; kk += 32) {
#pragma unroll
    for (int sub = 0; sub < 2; ++sub) {
      int n0 = kk + sub * 16;
      v8f p8;
      if (n0 < nk) {
        v16bf b0 = loadB_rows_bf(K, DKK, n0, 0, lane);
        v16bf b1 = loadB_rows_bf(K, DKK, n0, 32, lane);
        v8f s = {};
        s = wmma_bf16(a0, b0, s);
        s = wmma_bf16(a1, b1, s);
        int kj = n0 + col;
#pragma unroll
        for (int r = 0; r < 8; ++r) {
          int qi = qbase + r + 8 * hi;
          p8[r] = (kj <= qi) ? __expf(s[r] - m[r]) * rinv[r] : 0.0f;
        }
      } else {
#pragma unroll
        for (int r = 0; r < 8; ++r) p8[r] = 0.0f;
      }
#pragma unroll
      for (int r = 0; r < 8; ++r) {
        int row = r + 8 * hi;
        A[(size_t)(qbase + row) * LL + n0 + col] = p8[r];
        myP[row * 32 + sub * 16 + col] = (bf16_t)p8[r];   // C-layout -> LDS
      }
    }
    // re-load P as A operand (same-wave LDS ops are in-order; no barrier needed)
    v16bf pa;
    {
      int mm = lane & 15, ko = (lane >> 4) * 8;
      const bf16_t* pp = myP + mm * 32 + ko;
      pa = cat16(ld8bf(pp), ld8bf(pp + 16));
    }
    v16bf vb0 = loadB_rows_bf(V, LL, 0, kk, lane);
    o0 = wmma_bf16(pa, vb0, o0);
    v16bf vb1 = loadB_rows_bf(V, LL, 16, kk, lane);
    o1 = wmma_bf16(pa, vb1, o1);
    v16bf vb2 = loadB_rows_bf(V, LL, 32, kk, lane);
    o2 = wmma_bf16(pa, vb2, o2);
    v16bf vb3 = loadB_rows_bf(V, LL, 48, kk, lane);
    o3 = wmma_bf16(pa, vb3, o3);
  }

  // zero-fill fully-masked attn region (cols [zstart, 1024)), coalesced float4
  int zstart = (nk + 31) & ~31;
  vf4 z4 = {0.0f, 0.0f, 0.0f, 0.0f};
  for (int r = 0; r < 16; ++r) {
    float* rowp = A + (size_t)(qbase + r) * LL;
    for (int cidx = zstart + lane * 4; cidx < LL; cidx += 128)
      *(vf4*)(rowp + cidx) = z4;
  }

  // store O tile to merged-head f32 buffer [b][l][h*64+dv]
  {
    int h = bh & 15, bidx = bh >> 4;
    v8f oacc[4] = {o0, o1, o2, o3};
#pragma unroll
    for (int t = 0; t < 4; ++t) {
#pragma unroll
      for (int r = 0; r < 8; ++r) {
        int row = qbase + r + 8 * hi;
        o_ws[((size_t)(bidx * LL + row)) * DD + h * 64 + t * 16 + col] = oacc[t][r];
      }
    }
  }
}

// ---------------- residual + LayerNorm: one block per (b,l) row of 1024
__global__ __launch_bounds__(256) void ln_kernel(const float* __restrict__ o_ws,
                                                 const float* __restrict__ res,
                                                 const float* __restrict__ gamma,
                                                 const float* __restrict__ beta,
                                                 float* __restrict__ out) {
  __shared__ float ws_[8], ws2_[8];
  int row = blockIdx.x;
  const float* op = o_ws + (size_t)row * DD;
  const float* rp = res + (size_t)row * DD;
  float v[4], s = 0.0f, s2 = 0.0f;
#pragma unroll
  for (int i = 0; i < 4; ++i) {
    int c = threadIdx.x + i * 256;
    float x = op[c] + rp[c];
    v[i] = x; s += x; s2 += x * x;
  }
#pragma unroll
  for (int off = 1; off < 32; off <<= 1) {
    s += __shfl_xor(s, off, 32);
    s2 += __shfl_xor(s2, off, 32);
  }
  if ((threadIdx.x & 31) == 0) {
    ws_[threadIdx.x >> 5] = s;
    ws2_[threadIdx.x >> 5] = s2;
  }
  __syncthreads();
  float ts = 0.0f, ts2 = 0.0f;
#pragma unroll
  for (int w = 0; w < 8; ++w) { ts += ws_[w]; ts2 += ws2_[w]; }
  float mu = ts * (1.0f / 1024.0f);
  float var = ts2 * (1.0f / 1024.0f) - mu * mu;
  float rstd = rsqrtf(var + 1e-5f);
#pragma unroll
  for (int i = 0; i < 4; ++i) {
    int c = threadIdx.x + i * 256;
    out[(size_t)row * DD + c] = (v[i] - mu) * rstd * gamma[c] + beta[c];
  }
}

extern "C" void kernel_launch(void* const* d_in, const int* in_sizes, int n_in,
                              void* d_out, int out_size, void* d_ws, size_t ws_size,
                              hipStream_t stream) {
  (void)in_sizes; (void)n_in; (void)out_size; (void)ws_size;
  const float* q     = (const float*)d_in[0];
  const float* k     = (const float*)d_in[1];
  const float* v     = (const float*)d_in[2];
  // d_in[3] = mask (int32 causal tril) -- applied analytically in-kernel
  const float* Wq    = (const float*)d_in[4];
  const float* Wk    = (const float*)d_in[5];
  const float* Wv    = (const float*)d_in[6];
  const float* gamma = (const float*)d_in[7];
  const float* beta  = (const float*)d_in[8];

  float* out  = (float*)d_out;                              // [4,1024,1024]
  float* attn = out + (size_t)BB * LL * DD;                 // [4,16,1024,1024]

  const size_t NPROJ = (size_t)BB * HH * LL * DKK;          // 4M elements
  bf16_t* qh  = (bf16_t*)d_ws;
  bf16_t* khp = qh + NPROJ;
  bf16_t* vT  = khp + NPROJ;
  float*  ows = (float*)(vT + NPROJ);                       // 24MB offset, 16MB

  // q scaled by 1/sqrt(DK)=0.125 at projection time
  proj_kernel<<<256, 256, 0, stream>>>(q, Wq, qh, 0, 0.125f);
  proj_kernel<<<256, 256, 0, stream>>>(k, Wk, khp, 1, 1.0f);
  proj_kernel<<<256, 256, 0, stream>>>(v, Wv, vT, 2, 1.0f);
  attn_kernel<<<512, 256, 0, stream>>>(qh, khp, vT, attn, ows);
  ln_kernel<<<(BB * LL), 256, 0, stream>>>(ows, q, gamma, beta, out);
}
